// RelationExtractionModule_72825465471611
// MI455X (gfx1250) — matrix operand order repaired and checked by reading, not compiled, not run on hardware
//
#include <hip/hip_runtime.h>

// ---------------------------------------------------------------------------
// Problem constants (match reference)
// ---------------------------------------------------------------------------
constexpr int Bn = 32, Dd = 768, Ll = 512;
constexpr int Hh = 1024, Rr = 20, Ww = 3;
constexpr int Pp = 4096, Ss = 16;
constexpr int Kconv = Dd * Ww; // 2304

typedef __bf16 bf16;
typedef __attribute__((ext_vector_type(16))) __bf16 v16bf;
typedef __attribute__((ext_vector_type(8)))  float  v8f;

union FragU { v16bf v; uint4 u[2]; };

// ---------------------------------------------------------------------------
// fp32 -> bf16 elementwise convert (weights)
// ---------------------------------------------------------------------------
__global__ void cvt_f32_bf16_k(const float* __restrict__ s, bf16* __restrict__ d, int n) {
    int i = blockIdx.x * blockDim.x + threadIdx.x;
    if (i < n) d[i] = (bf16)s[i];
}

// ---------------------------------------------------------------------------
// im2col for width-3 "same" conv, position-major output:
//   xcol[b][n][kk] = x[b][kk/3][n + kk%3 - 1]  (zero padded), bf16
// One block per (b, n) row of 2304 contiguous elements.
// ---------------------------------------------------------------------------
__global__ void im2col_k(const float* __restrict__ x, bf16* __restrict__ xcol) {
    int row = blockIdx.x;            // b * Ll + n
    int b = row / Ll, n = row - b * Ll;
    const float* xb = x + (size_t)b * Dd * Ll;
    bf16* out = xcol + (size_t)row * Kconv;
    for (int kk = threadIdx.x; kk < Kconv; kk += blockDim.x) {
        int d = kk / 3;
        int w = kk - 3 * d;
        int p = n + w - 1;
        float v = (p >= 0 && p < Ll) ? xb[(size_t)d * Ll + p] : 0.0f;
        out[kk] = (bf16)v;
    }
}

// ---------------------------------------------------------------------------
// Generic bf16 WMMA GEMM:  C[m][n] = epilogue( sum_k A[m][k] * Bm[n][k] )
//   A : M x K row-major bf16   Bm : N x K row-major bf16
//   epilogue: + row_bias[m] + col_bias[n], ReLU, + res[m][n], store f32 (+bf16)
//
// Block = 256 threads = 8 waves; block tile 256(M) x 64(N); wave tile 32 x 64:
//   2 A-fragments x 4 B-fragments -> 8 v_wmma per k-step vs 12 b128 loads.
// Register double-buffering with a rotated, branch-free steady-state loop.
// __launch_bounds__(256, 1): allow the full per-lane VGPR budget (the kernel
// needs ~64 acc + 96 fragment VGPRs) so nothing spills to scratch; 8 fully
// independent accumulators give the matrix pipe plenty of in-wave ILP.
//
// Fragment layouts (16-bit, wave32):
//   A 16x32: lane m=l&15, khalf=l>>4 -> contiguous 8-elem runs at
//            k = kb+khalf*8 and kb+16+khalf*8.
//   B 32x16: lane n=l&15, khalf=l>>4 -> contiguous 8-elem runs at
//            k = kb+khalf*8 and kb+16+khalf*8.
// ---------------------------------------------------------------------------
__global__ void __launch_bounds__(256, 1)
gemm_wmma_k(const bf16* __restrict__ A,
            const bf16* __restrict__ Bm,
            float* __restrict__ C,
            bf16* __restrict__ Cb,
            const float* __restrict__ row_bias,
            const float* __restrict__ col_bias,
            const float* __restrict__ res,
            int K, int ldc,
            long strideB, long strideC, int relu) {
    const int lane = threadIdx.x & 31;
    const int wave = threadIdx.x >> 5;
    const int m0w = blockIdx.y * 256 + wave * 32;   // wave owns rows m0w..m0w+31
    const int n0  = blockIdx.x * 64;

    const bf16*  Bz  = Bm + (size_t)blockIdx.z * strideB;
    float*       Cz  = C  + (size_t)blockIdx.z * strideC;
    const float* Rz  = res ? res + (size_t)blockIdx.z * strideC : nullptr;
    bf16*        Cbz = Cb ? Cb + (size_t)blockIdx.z * strideC : nullptr;

    const int khalf = lane >> 4;
    const int lid   = lane & 15;

    const bf16* arow[2];
    arow[0] = A + (size_t)(m0w + lid) * K + khalf * 8;
    arow[1] = arow[0] + (size_t)16 * K;

    const bf16* brow[4];
#pragma unroll
    for (int t = 0; t < 4; ++t)
        brow[t] = Bz + (size_t)(n0 + t * 16 + lid) * K + khalf * 8;

    v8f acc[2][4] = {};

    FragU a0[2], a1[2], b0[4], b1[4];

    auto loadFrags = [&](FragU a[2], FragU b[4], int kb) {
#pragma unroll
        for (int ma = 0; ma < 2; ++ma) {
            a[ma].u[0] = *(const uint4*)(arow[ma] + kb);
            a[ma].u[1] = *(const uint4*)(arow[ma] + kb + 16);
        }
#pragma unroll
        for (int t = 0; t < 4; ++t) {
            const bf16* bp = brow[t] + kb;
            b[t].u[0] = *(const uint4*)(bp);
            b[t].u[1] = *(const uint4*)(bp + 16);
        }
    };
    auto compute = [&](FragU a[2], FragU b[4]) {
#pragma unroll
        for (int ma = 0; ma < 2; ++ma)
#pragma unroll
            for (int t = 0; t < 4; ++t)
                acc[ma][t] = __builtin_amdgcn_wmma_f32_16x16x32_bf16(
                    false, a[ma].v, false, b[t].v, (short)0, acc[ma][t],
                    false, false);
    };

    // branch-free software pipeline over pairs of 32-wide k-steps (K % 64 == 0)
    loadFrags(a0, b0, 0);
    int kb = 0;
    for (; kb + 64 < K; kb += 64) {
        loadFrags(a1, b1, kb + 32);      // prefetch while matrix pipe runs a0/b0
        compute(a0, b0);
        loadFrags(a0, b0, kb + 64);      // prefetch while matrix pipe runs a1/b1
        compute(a1, b1);
    }
    loadFrags(a1, b1, kb + 32);          // drain: last two k-steps
    compute(a0, b0);
    compute(a1, b1);

    // C/D layout: VGPR v holds M = 8*(lane>>4) + v, N = lane & 15
#pragma unroll
    for (int ma = 0; ma < 2; ++ma) {
        const int rbase = m0w + ma * 16 + khalf * 8;
#pragma unroll
        for (int t = 0; t < 4; ++t) {
            const int col = n0 + t * 16 + lid;
            const float cb = col_bias ? col_bias[col] : 0.0f;
#pragma unroll
            for (int v = 0; v < 8; ++v) {
                const int row = rbase + v;
                float val = acc[ma][t][v] + cb + (row_bias ? row_bias[row] : 0.0f);
                if (relu) val = fmaxf(val, 0.0f);
                const size_t o = (size_t)row * ldc + col;
                if (Rz) val += Rz[o];
                Cz[o] = val;
                if (Cbz) Cbz[o] = (bf16)val;
            }
        }
    }
}

// ---------------------------------------------------------------------------
// Span max-pool over (B, D, L) fp32 for head & tail spans; maxed = hmax+tmax,
// written bf16 (P x D) as the A operand of lin1.  One block per pair.
// ---------------------------------------------------------------------------
__global__ void spanmax_k(const float* __restrict__ y,
                          const int* __restrict__ pb,
                          const int* __restrict__ hidx, const int* __restrict__ hspn,
                          const int* __restrict__ tidx, const int* __restrict__ tspn,
                          bf16* __restrict__ maxed) {
    const int p = blockIdx.x;
    const int b = pb[p];
    const int h0 = hidx[p], hn = hspn[p];
    const int t0 = tidx[p], tn = tspn[p];
    const float* yb = y + (size_t)b * Dd * Ll;
    for (int d = threadIdx.x; d < Dd; d += blockDim.x) {
        const float* row = yb + (size_t)d * Ll;
        float hm = -__builtin_inff(), tm = -__builtin_inff();
#pragma unroll
        for (int j = 0; j < Ss; ++j) {
            int ph = min(h0 + j, Ll - 1);
            int pt = min(t0 + j, Ll - 1);
            if (j < hn) hm = fmaxf(hm, row[ph]);
            if (j < tn) tm = fmaxf(tm, row[pt]);
        }
        maxed[(size_t)p * Dd + d] = (bf16)(hm + tm);
    }
}

// ---------------------------------------------------------------------------
// Final head: logits[p][r] = h2[p] . rel_w[r] + rel_b[r]  (N=20, tiny -> VALU)
// Also copies labels into the tail of d_out.
// ---------------------------------------------------------------------------
__global__ void rel_k(const float* __restrict__ h2,
                      const float* __restrict__ rw, const float* __restrict__ rb,
                      const int* __restrict__ labels, float* __restrict__ out) {
    const int idx = blockIdx.x * blockDim.x + threadIdx.x;
    if (idx < Pp * Rr) {
        const int p = idx / Rr, r = idx - p * Rr;
        const float* hr = h2 + (size_t)p * Hh;
        const float* wr = rw + (size_t)r * Hh;
        float s = rb[r];
        for (int k = 0; k < Hh; ++k) s = fmaf(hr[k], wr[k], s);
        out[idx] = s;
    }
    if (idx < Pp) out[(size_t)Pp * Rr + idx] = (float)labels[idx];
}

// ---------------------------------------------------------------------------
// Host launcher
// ---------------------------------------------------------------------------
extern "C" void kernel_launch(void* const* d_in, const int* in_sizes, int n_in,
                              void* d_out, int out_size, void* d_ws, size_t ws_size,
                              hipStream_t stream) {
    (void)in_sizes; (void)n_in; (void)out_size; (void)ws_size;

    const float* x       = (const float*)d_in[0];
    const int*   pbatch  = (const int*)d_in[1];
    const int*   hidx    = (const int*)d_in[2];
    const int*   hspn    = (const int*)d_in[3];
    const int*   tidx    = (const int*)d_in[4];
    const int*   tspn    = (const int*)d_in[5];
    const int*   labels  = (const int*)d_in[6];
    const float* c1w     = (const float*)d_in[7];
    const float* c1b     = (const float*)d_in[8];
    const float* c2w     = (const float*)d_in[9];
    const float* c2b     = (const float*)d_in[10];
    const float* l1w     = (const float*)d_in[11];
    const float* l1b     = (const float*)d_in[12];
    const float* l2w     = (const float*)d_in[13];
    const float* l2b     = (const float*)d_in[14];
    const float* rw      = (const float*)d_in[15];
    const float* rb      = (const float*)d_in[16];
    float* out = (float*)d_out;

    // workspace carve-up (256B aligned)
    char* ws = (char*)d_ws;
    size_t off = 0;
    auto alloc = [&](size_t bytes) {
        char* p = ws + off;
        off = (off + bytes + 255) & ~(size_t)255;
        return p;
    };
    bf16*  xcol = (bf16*)alloc((size_t)Bn * Ll * Kconv * sizeof(bf16)); // 72 MB
    bf16*  w1b  = (bf16*)alloc((size_t)Dd * Kconv * sizeof(bf16));
    bf16*  w2b  = (bf16*)alloc((size_t)Dd * Kconv * sizeof(bf16));
    bf16*  w3b  = (bf16*)alloc((size_t)Hh * Dd * sizeof(bf16));
    bf16*  w4b  = (bf16*)alloc((size_t)Hh * Hh * sizeof(bf16));
    float* y1   = (float*)alloc((size_t)Bn * Dd * Ll * sizeof(float));  // 48 MB
    float* y2   = (float*)alloc((size_t)Bn * Dd * Ll * sizeof(float));  // 48 MB
    bf16*  mx   = (bf16*)alloc((size_t)Pp * Dd * sizeof(bf16));
    float* h1f  = (float*)alloc((size_t)Pp * Hh * sizeof(float));
    bf16*  h1b  = (bf16*)alloc((size_t)Pp * Hh * sizeof(bf16));
    float* h2f  = (float*)alloc((size_t)Pp * Hh * sizeof(float));

    const int TPB = 256;
    auto cvt = [&](const float* s, bf16* d, int n) {
        cvt_f32_bf16_k<<<(n + TPB - 1) / TPB, TPB, 0, stream>>>(s, d, n);
    };
    // weights -> bf16
    cvt(c1w, w1b, Dd * Kconv);
    cvt(c2w, w2b, Dd * Kconv);
    cvt(l1w, w3b, Hh * Dd);
    cvt(l2w, w4b, Hh * Hh);

    const long sB  = (long)Ll * Kconv;   // xcol per-batch stride (elements)
    const long sC  = (long)Dd * Ll;      // y per-batch stride

    // conv1: y1 = relu(W1 * im2col(x) + b1) + x
    im2col_k<<<Bn * Ll, TPB, 0, stream>>>(x, xcol);
    {
        dim3 g(Ll / 64, Dd / 256, Bn);
        gemm_wmma_k<<<g, TPB, 0, stream>>>(w1b, xcol, y1, nullptr,
                                           c1b, nullptr, x,
                                           Kconv, Ll, sB, sC, 1);
    }
    // conv2: y2 = relu(W2 * im2col(y1) + b2) + y1
    im2col_k<<<Bn * Ll, TPB, 0, stream>>>(y1, xcol);
    {
        dim3 g(Ll / 64, Dd / 256, Bn);
        gemm_wmma_k<<<g, TPB, 0, stream>>>(w2b, xcol, y2, nullptr,
                                           c2b, nullptr, y1,
                                           Kconv, Ll, sB, sC, 1);
    }
    // span max-pool -> maxed (bf16, P x D)
    spanmax_k<<<Pp, TPB, 0, stream>>>(y2, pbatch, hidx, hspn, tidx, tspn, mx);

    // lin1: h1 = relu(maxed @ W1^T + b)  (M=P, N=H, K=D)
    {
        dim3 g(Hh / 64, Pp / 256, 1);
        gemm_wmma_k<<<g, TPB, 0, stream>>>(mx, w3b, h1f, h1b,
                                           nullptr, l1b, nullptr,
                                           Dd, Hh, 0, 0, 1);
    }
    // lin2: h2 = relu(h1 @ W2^T + b)  (M=P, N=H, K=H)
    {
        dim3 g(Hh / 64, Pp / 256, 1);
        gemm_wmma_k<<<g, TPB, 0, stream>>>(h1b, w4b, h2f, nullptr,
                                           nullptr, l2b, nullptr,
                                           Hh, Hh, 0, 0, 1);
    }
    // rel head + labels
    rel_k<<<(Pp * Rr + TPB - 1) / TPB, TPB, 0, stream>>>(h2f, rw, rb, labels, out);
}
